// MengerWavKAN_26233660244250
// MI455X (gfx1250) — compile-verified, matrix operand order checked by reading, not run on hardware
//
#include <hip/hip_runtime.h>
#include <cmath>

typedef __bf16 bf16_t;
typedef __attribute__((ext_vector_type(16))) __bf16 v16bf;
typedef __attribute__((ext_vector_type(8)))  float  v8f;
typedef __attribute__((ext_vector_type(4)))  int    v4i;

// ---------------- constants ----------------
#define NROWS 16384
#define DDIM  768
#define HDIM  512
#define MDIM  256
#define ZDIM  256
#define KBR   20
#define EEDGE 262144
#define KP_E  800   // 794 packed enc coeffs padded to 25*32
#define KP_D  544   // 538 packed dec coeffs padded to 17*32

// db4 decomposition filters (pywt)
__device__ const float c_dec_lo[8] = {
  -0.010597401784997278f, 0.032883011666982945f, 0.030841381835986965f,
  -0.18703481171888114f, -0.02798376941698385f, 0.6308807679295904f,
   0.7148465705525415f,   0.23037781330885523f };
__device__ const float c_dec_hi[8] = {
  -0.23037781330885523f, 0.7148465705525415f, -0.6308807679295904f,
  -0.02798376941698385f, 0.18703481171888114f, 0.030841381835986965f,
  -0.032883011666982945f, -0.010597401784997278f };

__device__ __forceinline__ float gelu_f(float x) {
  return 0.5f * x * (1.0f + erff(x * 0.7071067811865476f));
}

// ---------------- async global->LDS copy (CDNA5 path, guarded) ----------------
#if defined(__has_builtin)
#if __has_builtin(__builtin_amdgcn_global_load_async_to_lds_b128)
#define HAVE_ASYNC_LDS 1
#endif
#endif

__device__ __forceinline__ void copy16(const bf16_t* g, bf16_t* l) {
#ifdef HAVE_ASYNC_LDS
  bf16_t* gn = const_cast<bf16_t*>(g);
  __builtin_amdgcn_global_load_async_to_lds_b128(
      (__attribute__((address_space(1))) v4i*)gn,
      (__attribute__((address_space(3))) v4i*)l, 0, 0);
#else
  *(uint4*)l = *(const uint4*)g;
#endif
}

__device__ __forceinline__ void async_join() {
#ifdef HAVE_ASYNC_LDS
#if __has_builtin(__builtin_amdgcn_s_wait_asynccnt)
  __builtin_amdgcn_s_wait_asynccnt(0);
#else
  asm volatile("s_wait_asynccnt 0x0" ::: "memory");
#endif
#endif
}

// ---------------- utility kernels ----------------
__global__ void zero_f32(float* p, size_t n) {
  size_t i = (size_t)blockIdx.x * blockDim.x + threadIdx.x;
  size_t st = (size_t)gridDim.x * blockDim.x;
  for (; i < n; i += st) p[i] = 0.0f;
}

__global__ void f32_to_bf16(const float* a, bf16_t* o, size_t n) {
  size_t i = (size_t)blockIdx.x * blockDim.x + threadIdx.x;
  size_t st = (size_t)gridDim.x * blockDim.x;
  for (; i < n; i += st) o[i] = (bf16_t)a[i];
}

// pack W_mu [Z,Z] and W_lv [Z,Z] side by side into bf16 [Z, 2Z]
__global__ void pack_wmulv(const float* wmu, const float* wlv, bf16_t* o) {
  size_t i = (size_t)blockIdx.x * blockDim.x + threadIdx.x;
  size_t st = (size_t)gridDim.x * blockDim.x;
  size_t n = (size_t)ZDIM * 2 * ZDIM;
  for (; i < n; i += st) {
    int k = (int)(i >> 9);
    int c = (int)(i & 511);
    float v = (c < ZDIM) ? wmu[k * ZDIM + c] : wlv[k * ZDIM + (c - ZDIM)];
    o[i] = (bf16_t)v;
  }
}

// ---------------- SpMM: xa = segment_sum(vals * x[cols], rows) ----------------
__global__ void spmm_scatter(const float* __restrict__ x, const int* __restrict__ rows,
                             const int* __restrict__ cols, const float* __restrict__ vals,
                             float* __restrict__ xa) {
  int e = blockIdx.x;
  int r = rows[e], c = cols[e];
  float v = vals[e];
  const float* src = x + (size_t)c * DDIM;
  float* dst = xa + (size_t)r * DDIM;
  for (int d = threadIdx.x; d < DDIM; d += blockDim.x)
    atomicAdd(&dst[d], v * src[d]);
}

// ---------------- 4-level db4 DWT (symmetric mode), packed bf16 out ----------------
// packed order per row: [cA4 | cD4 | cD3 | cD2 | cD1 | zero-pad to kpad]
__global__ void dwt4_pack(const float* __restrict__ in, bf16_t* __restrict__ out,
                          int n0, int kpad) {
  __shared__ float sa[768];
  __shared__ float st[400];
  int row = blockIdx.x, tid = threadIdx.x, bd = blockDim.x;
  const float* p = in + (size_t)row * n0;
  for (int i = tid; i < n0; i += bd) sa[i] = p[i];
  int l1 = (n0 + 7) >> 1, l2 = (l1 + 7) >> 1, l3 = (l2 + 7) >> 1, l4 = (l3 + 7) >> 1;
  int doff[4] = { l4 + l4 + l3 + l2, l4 + l4 + l3, l4 + l4, l4 };
  int total = l4 + l4 + l3 + l2 + l1;
  bf16_t* orow = out + (size_t)row * kpad;
  int cur = n0;
  for (int lvl = 0; lvl < 4; lvl++) {
    int on = (cur + 7) >> 1;
    __syncthreads();
    for (int o = tid; o < on; o += bd) {
      float lo = 0.f, hi = 0.f;
#pragma unroll
      for (int l = 0; l < 8; l++) {
        int idx = 2 * o + l - 6;               // xe[i] = x[sym(i-6)]
        if (idx < 0) idx = -idx - 1;
        if (idx >= cur) idx = 2 * cur - 1 - idx;
        float v = sa[idx];
        lo += v * c_dec_lo[7 - l];
        hi += v * c_dec_hi[7 - l];
      }
      st[o] = lo;
      orow[doff[lvl] + o] = (bf16_t)hi;
    }
    __syncthreads();
    for (int o = tid; o < on; o += bd) sa[o] = st[o];
    cur = on;
  }
  __syncthreads();
  for (int o = tid; o < cur; o += bd) orow[o] = (bf16_t)sa[o];
  for (int c = total + tid; c < kpad; c += bd) orow[c] = (bf16_t)0.0f;
}

// pack per-scale basis slices into dense bf16 B [kpad, ncols]
__global__ void pack_basis(const float* __restrict__ basis, bf16_t* __restrict__ out,
                           int n0, int ncols, int kpad) {
  int r = blockIdx.x, tid = threadIdx.x, bd = blockDim.x;
  int l1 = (n0 + 7) >> 1, l2 = (l1 + 7) >> 1, l3 = (l2 + 7) >> 1, l4 = (l3 + 7) >> 1;
  int segoff[5] = { 0, l4, 2 * l4, 2 * l4 + l3, 2 * l4 + l3 + l2 };
  int seglen[5] = { l4, l4, l3, l2, l1 };
  int total = 2 * l4 + l3 + l2 + l1;
  bf16_t* orow = out + (size_t)r * ncols;
  if (r >= total) {
    for (int c = tid; c < ncols; c += bd) orow[c] = (bf16_t)0.0f;
    return;
  }
  int s = 4;
#pragma unroll
  for (int i = 0; i < 5; i++)
    if (r >= segoff[i] && r < segoff[i] + seglen[i]) { s = i; break; }
  const float* src = basis + ((size_t)s * n0 + (r - segoff[s])) * ncols;
  for (int c = tid; c < ncols; c += bd) orow[c] = (bf16_t)src[c];
}

// ---------------- bf16 WMMA GEMM: C[f32] = A @ B ----------------
// 128 threads = 4 waves; block tile 64x64; wave tile 32x32 (2x2 WMMA, frag reuse);
// double-buffered LDS with async global->LDS copies.
__global__ void gemm_bf16(const bf16_t* __restrict__ A, const bf16_t* __restrict__ B,
                          float* __restrict__ C, int Ndim, int Kdim,
                          int lda, int ldb, int ldc) {
  __shared__ __attribute__((aligned(16))) bf16_t sA[2][64 * 32];  // [m][k]
  __shared__ __attribute__((aligned(16))) bf16_t sB[2][32 * 64];  // [k][n]
  int tid = threadIdx.x, wave = tid >> 5, lane = tid & 31;
  int m0 = blockIdx.x * 64, n0 = blockIdx.y * 64;
  int wm = (wave >> 1) * 32, wn = (wave & 1) * 32;
  int lm = lane & 15, kb = (lane < 16) ? 0 : 8, b8 = (lane < 16) ? 0 : 8;
  int nchunks = Kdim / 32;

  auto stage = [&](int c, int buf) {
    int kt = c * 32;
#pragma unroll
    for (int s = 0; s < 2; s++) {
      int i = tid * 8 + s * 1024;
      int r = i >> 5, cc = i & 31;
      copy16(A + (size_t)(m0 + r) * lda + kt + cc, &sA[buf][i]);
    }
#pragma unroll
    for (int s = 0; s < 2; s++) {
      int i = tid * 8 + s * 1024;
      int k = i >> 6, nn = i & 63;
      copy16(B + (size_t)(kt + k) * ldb + n0 + nn, &sB[buf][i]);
    }
  };

  stage(0, 0);
  async_join();
  __syncthreads();

  v8f acc[2][2];
#pragma unroll
  for (int i = 0; i < 2; i++)
#pragma unroll
    for (int j = 0; j < 2; j++) acc[i][j] = (v8f){};

  for (int c = 0; c < nchunks; c++) {
    int cur = c & 1;
    if (c + 1 < nchunks) stage(c + 1, cur ^ 1);
    const bf16_t* pa = sA[cur];
    const bf16_t* pb = sB[cur];
    v16bf af[2], bf[2];
#pragma unroll
    for (int i = 0; i < 2; i++) {
      int mrow = wm + i * 16 + lm;
#pragma unroll
      for (int e = 0; e < 16; e++) {
        int kk = (e < 8) ? (kb + e) : (kb + 8 + e);   // ISA 16-bit K split per lane half
        af[i][e] = pa[mrow * 32 + kk];
      }
    }
#pragma unroll
    for (int j = 0; j < 2; j++) {
      int ncol = wn + j * 16 + lm;
#pragma unroll
      for (int e = 0; e < 16; e++) {
        int kk = (e < 8) ? (kb + e) : (kb + 8 + e);
        bf[j][e] = pb[kk * 64 + ncol];
      }
    }
#pragma unroll
    for (int i = 0; i < 2; i++)
#pragma unroll
      for (int j = 0; j < 2; j++)
        acc[i][j] = __builtin_amdgcn_wmma_f32_16x16x32_bf16(
            false, af[i], false, bf[j], (short)0, acc[i][j], false, false);
    async_join();
    __syncthreads();
  }

#pragma unroll
  for (int i = 0; i < 2; i++) {
    int gm = m0 + wm + i * 16 + b8;
#pragma unroll
    for (int j = 0; j < 2; j++) {
      int gn = n0 + wn + j * 16 + lm;
#pragma unroll
      for (int r = 0; r < 8; r++)
        C[(size_t)(gm + r) * ldc + gn] = acc[i][j][r];
    }
  }
}

// ---------------- fused dendritic layer: max_k GELU(A @ W[k] + b[k]) ----------------
// 128 threads = 4 waves; block tile 32x64; A strip (32 x Kdim <= 32KB) cached in LDS;
// branch weights streamed through async double buffer (flattened branch*chunk loop).
__global__ void dendritic_bf16(const bf16_t* __restrict__ A, const bf16_t* __restrict__ W,
                               const float* __restrict__ bias, void* __restrict__ out,
                               int out_is_bf16, int Ndim, int Kdim, int Kbr,
                               int lda, int ldc) {
  __shared__ __attribute__((aligned(16))) bf16_t sA[32 * 512];
  __shared__ __attribute__((aligned(16))) bf16_t sB[2][32 * 64];  // [k][n]
  int tid = threadIdx.x, wave = tid >> 5, lane = tid & 31;
  int m0 = blockIdx.x * 32, n0 = blockIdx.y * 64;
  int wm = (wave >> 1) * 16, wn = (wave & 1) * 32;
  int lm = lane & 15, kb = (lane < 16) ? 0 : 8, b8 = (lane < 16) ? 0 : 8;
  int cpb = Kdim / 32, total = Kbr * cpb;

  // cache A strip once
  for (int i = tid * 8; i < 32 * Kdim; i += 128 * 8) {
    int r = i / Kdim, c = i % Kdim;
    copy16(A + (size_t)(m0 + r) * lda + c, &sA[i]);
  }

  auto stageB = [&](int c, int buf) {
    int br = c / cpb, kt = (c % cpb) * 32;
    const bf16_t* Wb = W + (size_t)br * Kdim * Ndim;
#pragma unroll
    for (int s = 0; s < 2; s++) {
      int i = tid * 8 + s * 1024;
      int k = i >> 6, nn = i & 63;
      copy16(Wb + (size_t)(kt + k) * Ndim + n0 + nn, &sB[buf][i]);
    }
  };

  stageB(0, 0);
  async_join();
  __syncthreads();

  float best[2][8];
#pragma unroll
  for (int j = 0; j < 2; j++)
#pragma unroll
    for (int r = 0; r < 8; r++) best[j][r] = -INFINITY;

  v8f acc[2];
  for (int c = 0; c < total; c++) {
    int cur = c & 1;
    if (c + 1 < total) stageB(c + 1, cur ^ 1);
    int kt = (c % cpb) * 32;
    if (c % cpb == 0) {
#pragma unroll
      for (int j = 0; j < 2; j++) acc[j] = (v8f){};
    }
    int mrow = wm + lm;
    v16bf af, bf[2];
#pragma unroll
    for (int e = 0; e < 16; e++) {
      int kk = (e < 8) ? (kb + e) : (kb + 8 + e);
      af[e] = sA[mrow * Kdim + kt + kk];
    }
    const bf16_t* pb = sB[cur];
#pragma unroll
    for (int j = 0; j < 2; j++) {
      int ncol = wn + j * 16 + lm;
#pragma unroll
      for (int e = 0; e < 16; e++) {
        int kk = (e < 8) ? (kb + e) : (kb + 8 + e);
        bf[j][e] = pb[kk * 64 + ncol];
      }
    }
#pragma unroll
    for (int j = 0; j < 2; j++)
      acc[j] = __builtin_amdgcn_wmma_f32_16x16x32_bf16(
          false, af, false, bf[j], (short)0, acc[j], false, false);
    if (c % cpb == cpb - 1) {
      int br = c / cpb;
#pragma unroll
      for (int j = 0; j < 2; j++) {
        float bs = bias[(size_t)br * Ndim + n0 + wn + j * 16 + lm];
#pragma unroll
        for (int r = 0; r < 8; r++) {
          float g = gelu_f(acc[j][r] + bs);
          best[j][r] = fmaxf(best[j][r], g);
        }
      }
    }
    async_join();
    __syncthreads();
  }

  int gm = m0 + wm + b8;
  if (out_is_bf16) {
    bf16_t* O = (bf16_t*)out;
#pragma unroll
    for (int j = 0; j < 2; j++) {
      int gn = n0 + wn + j * 16 + lm;
#pragma unroll
      for (int r = 0; r < 8; r++)
        O[(size_t)(gm + r) * ldc + gn] = (bf16_t)best[j][r];
    }
  } else {
    float* O = (float*)out;
#pragma unroll
    for (int j = 0; j < 2; j++) {
      int gn = n0 + wn + j * 16 + lm;
#pragma unroll
      for (int r = 0; r < 8; r++)
        O[(size_t)(gm + r) * ldc + gn] = best[j][r];
    }
  }
}

// ---------------- LayerNorm (f32 in -> bf16 out) ----------------
__global__ void layernorm_b(const float* __restrict__ in, const float* __restrict__ g,
                            const float* __restrict__ b, bf16_t* __restrict__ out, int H) {
  __shared__ float red[256];
  int row = blockIdx.x, tid = threadIdx.x;
  const float* p = in + (size_t)row * H;
  float s = 0.f;
  for (int i = tid; i < H; i += 256) s += p[i];
  red[tid] = s; __syncthreads();
  for (int st = 128; st > 0; st >>= 1) { if (tid < st) red[tid] += red[tid + st]; __syncthreads(); }
  float mean = red[0] / (float)H;
  __syncthreads();
  float v = 0.f;
  for (int i = tid; i < H; i += 256) { float d = p[i] - mean; v += d * d; }
  red[tid] = v; __syncthreads();
  for (int st = 128; st > 0; st >>= 1) { if (tid < st) red[tid] += red[tid + st]; __syncthreads(); }
  float rstd = rsqrtf(red[0] / (float)H + 1e-5f);
  for (int i = tid; i < H; i += 256)
    out[(size_t)row * H + i] = (bf16_t)((p[i] - mean) * rstd * g[i] + b[i]);
}

// ---------------- reparameterization; splits muv into mu/logvar (-> d_out) + z (bf16) ----------------
__global__ void z_split(const float* __restrict__ muv, const float* __restrict__ bmu,
                        const float* __restrict__ blv, const float* __restrict__ eps,
                        float* __restrict__ mu_o, float* __restrict__ lv_o,
                        bf16_t* __restrict__ zb) {
  size_t i = (size_t)blockIdx.x * blockDim.x + threadIdx.x;
  size_t n = (size_t)NROWS * ZDIM;
  if (i >= n) return;
  size_t row = i / ZDIM; int j = (int)(i % ZDIM);
  float mu = muv[row * (2 * ZDIM) + j] + bmu[j];
  float lv = muv[row * (2 * ZDIM) + ZDIM + j] + blv[j];
  mu_o[i] = mu;
  lv_o[i] = lv;
  zb[i] = (bf16_t)(mu + eps[i] * expf(0.5f * lv));
}

// ---------------- launch ----------------
static inline size_t au(size_t v) { return (v + 255) & ~(size_t)255; }

extern "C" void kernel_launch(void* const* d_in, const int* in_sizes, int n_in,
                              void* d_out, int out_size, void* d_ws, size_t ws_size,
                              hipStream_t stream) {
  const float* x        = (const float*)d_in[0];
  const int*   arows    = (const int*)d_in[1];
  const int*   acols    = (const int*)d_in[2];
  const float* avals    = (const float*)d_in[3];
  const float* basis_e  = (const float*)d_in[4];
  const float* ln_g     = (const float*)d_in[5];
  const float* ln_b     = (const float*)d_in[6];
  const float* Wd_enc   = (const float*)d_in[7];
  const float* bd_enc   = (const float*)d_in[8];
  const float* W_mu     = (const float*)d_in[9];
  const float* b_mu     = (const float*)d_in[10];
  const float* W_lv     = (const float*)d_in[11];
  const float* b_lv     = (const float*)d_in[12];
  const float* Wd_dec   = (const float*)d_in[13];
  const float* bd_dec   = (const float*)d_in[14];
  const float* basis_d  = (const float*)d_in[15];
  const float* eps      = (const float*)d_in[16];

  float* out_recon = (float*)d_out;                                   // [N, D]
  float* out_mu    = out_recon + (size_t)NROWS * DDIM;                // [N, Z]
  float* out_lv    = out_mu    + (size_t)NROWS * ZDIM;                // [N, Z]

  char* w = (char*)d_ws; size_t o = 0;
  float*  xa   = (float*)(w + o);  o += au((size_t)NROWS * DDIM * 4);
  bf16_t* ce   = (bf16_t*)(w + o); o += au((size_t)NROWS * KP_E * 2);
  bf16_t* bpe  = (bf16_t*)(w + o); o += au((size_t)KP_E * HDIM * 2);
  float*  hf   = (float*)(w + o);  o += au((size_t)NROWS * HDIM * 4);  // reused for muv [N,2Z]
  bf16_t* hb   = (bf16_t*)(w + o); o += au((size_t)NROWS * HDIM * 2);
  bf16_t* wde  = (bf16_t*)(w + o); o += au((size_t)KBR * HDIM * MDIM * 2);
  bf16_t* encb = (bf16_t*)(w + o); o += au((size_t)NROWS * MDIM * 2);
  bf16_t* wmlv = (bf16_t*)(w + o); o += au((size_t)ZDIM * 2 * ZDIM * 2);
  bf16_t* zb   = (bf16_t*)(w + o); o += au((size_t)NROWS * ZDIM * 2);
  bf16_t* wdd  = (bf16_t*)(w + o); o += au((size_t)KBR * ZDIM * HDIM * 2);
  bf16_t* bpd  = (bf16_t*)(w + o); o += au((size_t)KP_D * DDIM * 2);
  float*  dec  = xa;   // xa dead after enc DWT
  bf16_t* cd   = ce;   // enc coeffs dead after KAN-enc GEMM

  // 1) xa = A @ x  (COO scatter)
  zero_f32<<<2048, 256, 0, stream>>>(xa, (size_t)NROWS * DDIM);
  spmm_scatter<<<EEDGE, 256, 0, stream>>>(x, arows, acols, avals, xa);

  // 2) wavelet-KAN encoder: packed DWT coeffs @ packed basis
  dwt4_pack<<<NROWS, 256, 0, stream>>>(xa, ce, DDIM, KP_E);
  pack_basis<<<KP_E, 256, 0, stream>>>(basis_e, bpe, DDIM, HDIM, KP_E);
  gemm_bf16<<<dim3(NROWS / 64, HDIM / 64), 128, 0, stream>>>(
      ce, bpe, hf, HDIM, KP_E, KP_E, HDIM, HDIM);

  // 3) LayerNorm -> bf16
  layernorm_b<<<NROWS, 256, 0, stream>>>(hf, ln_g, ln_b, hb, HDIM);

  // 4) dendritic encoder: max_k GELU(h @ Wd_enc[k] + bd_enc[k]) -> bf16
  f32_to_bf16<<<2048, 256, 0, stream>>>(Wd_enc, wde, (size_t)KBR * HDIM * MDIM);
  dendritic_bf16<<<dim3(NROWS / 32, MDIM / 64), 128, 0, stream>>>(
      hb, wde, bd_enc, encb, 1, MDIM, HDIM, KBR, HDIM, MDIM);

  // 5) mu/logvar in one GEMM (bias folded into z_split), then reparameterize
  pack_wmulv<<<512, 256, 0, stream>>>(W_mu, W_lv, wmlv);
  gemm_bf16<<<dim3(NROWS / 64, (2 * ZDIM) / 64), 128, 0, stream>>>(
      encb, wmlv, hf, 2 * ZDIM, ZDIM, MDIM, 2 * ZDIM, 2 * ZDIM);
  z_split<<<(NROWS * ZDIM + 255) / 256, 256, 0, stream>>>(
      hf, b_mu, b_lv, eps, out_mu, out_lv, zb);

  // 6) dendritic decoder -> f32
  f32_to_bf16<<<2048, 256, 0, stream>>>(Wd_dec, wdd, (size_t)KBR * ZDIM * HDIM);
  dendritic_bf16<<<dim3(NROWS / 32, HDIM / 64), 128, 0, stream>>>(
      zb, wdd, bd_dec, dec, 0, HDIM, ZDIM, KBR, ZDIM, HDIM);

  // 7) wavelet-KAN decoder -> recon directly into d_out
  dwt4_pack<<<NROWS, 256, 0, stream>>>(dec, cd, HDIM, KP_D);
  pack_basis<<<KP_D, 256, 0, stream>>>(basis_d, bpd, HDIM, DDIM, KP_D);
  gemm_bf16<<<dim3(NROWS / 64, DDIM / 64), 128, 0, stream>>>(
      cd, bpd, out_recon, DDIM, KP_D, KP_D, DDIM, DDIM);
}
